// Kernel_72739566125068
// MI455X (gfx1250) — matrix operand order repaired and checked by reading, not compiled, not run on hardware
//
#include <hip/hip_runtime.h>
#include <math.h>

typedef __attribute__((ext_vector_type(16))) _Float16 v16h;
typedef __attribute__((ext_vector_type(8)))  float    v8f;

#define NS 256
#define NT 16384
#define HCH 64
#define EPSQ 1e-16f          // EPS^2, EPS = 1e-8
#define HALF_LN2 0.34657359027997264f

// ---------------------------------------------------------------------------
// Prep kernel: fuse x = feat*w1 + b1 into the Pade polynomials so the hot
// loop evaluates degree-2 polynomials directly in `feat`.
//   num(feat) = al0 + al1*f + al2*f^2
//   den(feat) = 1 + | be0 + be1*f + be2*f^2 |
// ws layout (floats): [c*8 + {al0,al1,al2,be0,be1,be2,w2[c][0],w2[c][1]}],
// ws[512] = ln(reference_length), ws[513] = sum(source_strengths)
// ---------------------------------------------------------------------------
__global__ void prep_consts(const float* __restrict__ refL,
                            const float* __restrict__ W1,
                            const float* __restrict__ b1,
                            const float* __restrict__ pa,
                            const float* __restrict__ pb,
                            const float* __restrict__ W2,
                            const float* __restrict__ q,
                            float* __restrict__ ws) {
    int c = threadIdx.x;
    if (c < HCH) {
        float w1 = W1[c], bb = b1[c];
        float a0 = pa[c*3+0], a1 = pa[c*3+1], a2 = pa[c*3+2];
        float p0 = pb[c*2+0], p1 = pb[c*2+1];
        ws[c*8+0] = a0 + bb*(a1 + a2*bb);      // al0
        ws[c*8+1] = w1*(a1 + 2.0f*a2*bb);      // al1
        ws[c*8+2] = a2*w1*w1;                  // al2
        ws[c*8+3] = bb*(p0 + p1*bb);           // be0
        ws[c*8+4] = w1*(p0 + 2.0f*p1*bb);      // be1
        ws[c*8+5] = p1*w1*w1;                  // be2
        ws[c*8+6] = W2[c*2+0];                 // w20 -> B col 0
        ws[c*8+7] = W2[c*2+1];                 // w21 -> VALU out1 dot
    }
    if (c == 0) {
        ws[512] = logf(refL[0]);
        float s = 0.0f;
        for (int i = 0; i < NS; ++i) s += q[i];
        ws[513] = s;
    }
}

// ---------------------------------------------------------------------------
// Main kernel. 256 threads = 8 waves; each wave owns 16 targets and loops over
// all 256 sources, 4 at a time.
//
// Lane mapping (wave32): m = lane&15 selects pair row M, half = lane>>4.
// Per the CDNA5 16-bit A-fragment layout, lane (half==0) holds K in
// {0..7,16..23} and lane (half==1) holds K in {8..15,24..31} of the same row,
// so each lane evaluates exactly the 32 channels its A fragments need.
//
// Two K-block passes keep only 4 A fragments (32 VGPRs) live at a time;
// WMMAs alternate accumulators C0/C1 so dependent WMMAs are >=2 apart and the
// hazard slots fill with the other pass's channel VALU. A sched_barrier at
// the end of each source group stops cross-group pipelining so allocation
// stays inside the 256-VGPR window (no s_set_vgpr_msb churn).
// ---------------------------------------------------------------------------
__global__ __launch_bounds__(256) __attribute__((amdgpu_num_vgpr(256)))
void fmm_pade_kernel(const float* __restrict__ tp,   // target_points [NT,3]
                     const float* __restrict__ sp,   // source_points [NS,3]
                     const float* __restrict__ q,    // strengths [NS]
                     const float* __restrict__ b2,   // [2]
                     const float* __restrict__ ws,   // prepped constants
                     float* __restrict__ out) {      // [NT,4]
    __shared__ float4 cst4[HCH*2];      // {al0,al1,al2,be0} | {be1,be2,w20,w21}
    __shared__ float4 lsrc[NS];         // {Sx,Sy,Sz,q} per source
    __shared__ float  sstage[8*16];     // per-wave scalar extraction

    int tid = threadIdx.x;
    for (int i = tid; i < HCH*2; i += 256)
        cst4[i] = ((const float4*)ws)[i];
    {   // stage sources: one per thread
        int s = tid;
        float4 v;
        v.x = sp[s*3+0]; v.y = sp[s*3+1]; v.z = sp[s*3+2]; v.w = q[s];
        lsrc[s] = v;
    }
    __syncthreads();

    const float lnL  = ws[512];
    const float sumq = ws[513];
    const float b2_0 = b2[0];
    const float b2_1 = b2[1];

    const int lane  = tid & 31;
    const int wave  = tid >> 5;
    const int m     = lane & 15;
    const int half_ = lane >> 4;
    const int tile  = (blockIdx.x * 8 + wave) * 16;
    const int t     = tile + m;

    const float Tx = tp[t*3+0];
    const float Ty = tp[t*3+1];
    const float Tz = tp[t*3+2];

    // Constant B fragments (32x16 f16; K split by lane half, col = lane&15;
    // cols 0/1 carry W2[:,0]/W2[:,1], cols 2..15 zero).
    v16h B1, B2;
    {
        const int col = m;
        const int kb  = half_ * 8;
        #pragma unroll
        for (int k = 0; k < 16; ++k) {
            float v1 = 0.0f, v2 = 0.0f;
            if (col < 2) {
                const float* c1 = (const float*)&cst4[(kb + ((k < 8) ? k : (k + 8))) * 2 + 1];
                const float* c2 = (const float*)&cst4[(32 + kb + ((k < 8) ? k : (k + 8))) * 2 + 1];
                v1 = c1[2 + col];   // .z = w20, .w = w21
                v2 = c2[2 + col];
            }
            B1[k] = (_Float16)v1;
            B2[k] = (_Float16)v2;
        }
    }

    v8f C0 = {};   // WMMA accumulators (scalar field in col 0), summed at end
    v8f C1 = {};
    float vx = 0.0f, vy = 0.0f, vz = 0.0f;

    for (int s0 = 0; s0 < NS; s0 += 4) {
        float feat[4], invd[4], qs[4], rx[4], ry[4], rz[4], o1[4];
        #pragma unroll
        for (int j = 0; j < 4; ++j) {
            const float4 S = lsrc[s0 + j];
            qs[j] = S.w;
            rx[j] = Tx - S.x;
            ry[j] = Ty - S.y;
            rz[j] = Tz - S.z;
            float d2 = fmaf(rx[j], rx[j],
                       fmaf(ry[j], ry[j],
                       fmaf(rz[j], rz[j], EPSQ)));
            // log(d/L) = 0.5*ln2*log2(d2) - ln(L)
            feat[j] = fmaf(HALF_LN2, __log2f(d2), -lnL);
            invd[j] = __builtin_amdgcn_rsqf(d2);
            o1[j]   = 0.0f;
        }

        // ---- pass 0: channels 0..31 (K-block 0) ----
        {
            v16h A[4];
            #pragma unroll
            for (int i = 0; i < 16; ++i) {
                const int ch = (half_ << 3) + ((i < 8) ? i : (i + 8));
                const float4 ca = cst4[ch*2+0];   // al0 al1 al2 be0
                const float4 cb = cst4[ch*2+1];   // be1 be2 w20 w21
                #pragma unroll
                for (int j = 0; j < 4; ++j) {
                    const float f   = feat[j];
                    const float num = fmaf(fmaf(ca.z, f, ca.y), f, ca.x);
                    const float p   = fmaf(fmaf(cb.y, f, cb.x), f, ca.w);
                    const float den = 1.0f + fabsf(p);
                    const float h   = num * __builtin_amdgcn_rcpf(den);
                    o1[j] = fmaf(h, cb.w, o1[j]);
                    A[j][i] = (_Float16)(h * qs[j]);
                }
            }
            C0 = __builtin_amdgcn_wmma_f32_16x16x32_f16(
                     false, A[0], false, B1, (short)0, C0, false, false);
            C1 = __builtin_amdgcn_wmma_f32_16x16x32_f16(
                     false, A[1], false, B1, (short)0, C1, false, false);
            C0 = __builtin_amdgcn_wmma_f32_16x16x32_f16(
                     false, A[2], false, B1, (short)0, C0, false, false);
            C1 = __builtin_amdgcn_wmma_f32_16x16x32_f16(
                     false, A[3], false, B1, (short)0, C1, false, false);
        }

        // ---- pass 1: channels 32..63 (K-block 1) ----
        {
            v16h A[4];
            #pragma unroll
            for (int i = 0; i < 16; ++i) {
                const int ch = 32 + (half_ << 3) + ((i < 8) ? i : (i + 8));
                const float4 ca = cst4[ch*2+0];
                const float4 cb = cst4[ch*2+1];
                #pragma unroll
                for (int j = 0; j < 4; ++j) {
                    const float f   = feat[j];
                    const float num = fmaf(fmaf(ca.z, f, ca.y), f, ca.x);
                    const float p   = fmaf(fmaf(cb.y, f, cb.x), f, ca.w);
                    const float den = 1.0f + fabsf(p);
                    const float h   = num * __builtin_amdgcn_rcpf(den);
                    o1[j] = fmaf(h, cb.w, o1[j]);
                    A[j][i] = (_Float16)(h * qs[j]);
                }
            }
            C0 = __builtin_amdgcn_wmma_f32_16x16x32_f16(
                     false, A[0], false, B2, (short)0, C0, false, false);
            C1 = __builtin_amdgcn_wmma_f32_16x16x32_f16(
                     false, A[1], false, B2, (short)0, C1, false, false);
            C0 = __builtin_amdgcn_wmma_f32_16x16x32_f16(
                     false, A[2], false, B2, (short)0, C0, false, false);
            C1 = __builtin_amdgcn_wmma_f32_16x16x32_f16(
                     false, A[3], false, B2, (short)0, C1, false, false);
        }

        // ---- vector-field epilogue (independent VALU; fills WMMA slots) ----
        #pragma unroll
        for (int j = 0; j < 4; ++j) {
            float tot = o1[j] + __shfl_xor(o1[j], 16, 32) + b2_1;
            float w   = tot * qs[j] * invd[j];
            vx = fmaf(w, rx[j], vx);
            vy = fmaf(w, ry[j], vy);
            vz = fmaf(w, rz[j], vz);
        }

        // Stop cross-group software pipelining: keeps the live set inside the
        // 256-VGPR window (no s_set_vgpr_msb per-instruction overhead).
        __builtin_amdgcn_sched_barrier(0);
    }

    // Combine accumulators; extract scalar (col 0 lives in lanes 0 and 16).
    v8f C;
    #pragma unroll
    for (int jj = 0; jj < 8; ++jj) C[jj] = C0[jj] + C1[jj];

    if (m == 0) {
        #pragma unroll
        for (int jj = 0; jj < 8; ++jj)
            sstage[wave*16 + half_*8 + jj] = C[jj];
    }
    asm volatile("s_wait_dscnt 0" ::: "memory");
    if (half_ == 0) {
        float scalar = sstage[wave*16 + m] + b2_0 * sumq;
        float4 o;
        o.x = scalar; o.y = vx; o.z = vy; o.w = vz;
        ((float4*)out)[t] = o;
    }
}

// ---------------------------------------------------------------------------
extern "C" void kernel_launch(void* const* d_in, const int* in_sizes, int n_in,
                              void* d_out, int out_size, void* d_ws, size_t ws_size,
                              hipStream_t stream) {
    const float* refL = (const float*)d_in[0];
    const float* sp   = (const float*)d_in[1];
    const float* tpnt = (const float*)d_in[2];
    const float* q    = (const float*)d_in[3];
    const float* W1   = (const float*)d_in[4];
    const float* b1   = (const float*)d_in[5];
    const float* pa   = (const float*)d_in[6];
    const float* pb   = (const float*)d_in[7];
    const float* W2   = (const float*)d_in[8];
    const float* b2   = (const float*)d_in[9];
    float* ws  = (float*)d_ws;
    float* out = (float*)d_out;

    prep_consts<<<1, 64, 0, stream>>>(refL, W1, b1, pa, pb, W2, q, ws);

    // 8 waves/block * 16 targets/wave = 128 targets per block
    const int blocks = NT / 128;   // 128
    fmm_pade_kernel<<<blocks, 256, 0, stream>>>(tpnt, sp, q, b2, ws, out);
}
